// ConvCBAM_64372969832914
// MI455X (gfx1250) — matrix operand order repaired and checked by reading, not compiled, not run on hardware
//
#include <hip/hip_runtime.h>
#include <hip/hip_bf16.h>
#include <math.h>

typedef __attribute__((ext_vector_type(16))) __bf16 v16bf;
typedef __attribute__((ext_vector_type(8)))  float  v8f;

#define B_    16
#define C_    18
#define L_    131072
#define HID_  72            // 4*C
#define NT1_  5             // ceil(72/16) N-tiles of GEMM1
#define NT2_  2             // ceil(18/16) N-tiles of GEMM2
#define KC2_  3             // ceil(72/32) K-chunks of GEMM2

union Frag32B { v16bf v; uint4 u[2]; };   // one 16x(16xbf16) wmma operand fragment per lane

// Branch-free exact-GELU: erf via Abramowitz-Stegun 7.1.26 (|err|<=1.5e-7),
// native v_rcp_f32 + v_exp_f32, no EXEC divergence (keeps WMMA region clean).
__device__ __forceinline__ float gelu_exact(float v) {
    float xs = v * 0.70710678118654752440f;
    float a  = fabsf(xs);
    float t  = __builtin_amdgcn_rcpf(fmaf(0.3275911f, a, 1.0f));
    float p  = t * fmaf(t, fmaf(t, fmaf(t, fmaf(t, 1.061405429f, -1.453152027f),
                                        1.421413741f), -0.284496736f), 0.254829592f);
    float e  = __expf(-a * a);
    float er = copysignf(fmaf(-p, e, 1.0f), xs);
    return 0.5f * v * (1.0f + er);
}

__device__ __forceinline__ void atomicMaxFloatGlobal(float* addr, float v) {
    if (v >= 0.0f) atomicMax((int*)addr, __float_as_int(v));
    else           atomicMin((unsigned int*)addr, __float_as_uint(v));
}
__device__ __forceinline__ void atomicMaxFloatShared(int* addr, float v) {
    if (v >= 0.0f) atomicMax(addr, __float_as_int(v));
    else           atomicMin((unsigned int*)addr, __float_as_uint(v));
}

// Depthwise conv (K=7, zero pad) + channel LayerNorm for one position.
// CHECK=false is the interior fast path (no per-tap bounds tests).
template<bool CHECK>
__device__ __forceinline__ void conv_ln_row(const float* __restrict__ x,
                                            const float* __restrict__ dw_w,
                                            const float* __restrict__ dw_b,
                                            const float* __restrict__ ln_w,
                                            const float* __restrict__ ln_b,
                                            int b, int l, __bf16* __restrict__ hrow)
{
    float h[C_];
    float mu = 0.0f;
    #pragma unroll
    for (int c = 0; c < C_; ++c) {
        const float* xr = x + ((size_t)b * C_ + c) * L_;
        float acc = dw_b[c];
        #pragma unroll
        for (int k = 0; k < 7; ++k) {
            int ll = l + k - 3;
            float xv;
            if (CHECK) xv = (ll >= 0 && ll < L_) ? xr[ll] : 0.0f;
            else       xv = xr[ll];
            acc = fmaf(xv, dw_w[c * 7 + k], acc);
        }
        h[c] = acc;
        mu += acc;
    }
    mu *= (1.0f / C_);
    float var = 0.0f;
    #pragma unroll
    for (int c = 0; c < C_; ++c) { float d = h[c] - mu; var = fmaf(d, d, var); }
    float rstd = rsqrtf(var * (1.0f / C_) + 1e-6f);
    #pragma unroll
    for (int c = 0; c < C_; ++c)
        hrow[c] = (__bf16)(fmaf((h[c] - mu) * rstd, ln_w[c], ln_b[c]));
    #pragma unroll
    for (int c = C_; c < 32; ++c) hrow[c] = (__bf16)0.0f;
}

// ---------------------------------------------------------------------------
// k0: pack weights into CDNA5 WMMA B-fragment lane layout; init reductions.
//   B frag (32x16 K x N, bf16): lane = {kh = lane>>4, n = lane&15},
//   element i (0..7) holds K = kh*8 + i, element i (8..15) holds K = 16 + kh*8 + (i-8), column N = n.
// ---------------------------------------------------------------------------
__global__ void pack_init_kernel(const float* __restrict__ w1, const float* __restrict__ w2,
                                 __bf16* __restrict__ b1f, __bf16* __restrict__ b2f,
                                 float* __restrict__ gsum, float* __restrict__ gmax)
{
    const int t = threadIdx.x;
    // GEMM1: B[k][n] = w1[n][k]   (K padded 18->32, N padded 72->80)
    for (int e = t; e < NT1_ * 32 * 16; e += 256) {
        int nt = e >> 9, rem = e & 511, lane = rem >> 4, i = rem & 15;
        int nl = lane & 15, kh = lane >> 4;
        int k = (i < 8) ? (kh * 8 + i) : (16 + kh * 8 + (i - 8));
        int n = nt * 16 + nl;
        float v = (k < C_ && n < HID_) ? w1[n * C_ + k] : 0.0f;
        b1f[e] = (__bf16)v;
    }
    // GEMM2: B[k][n] = w2[n][k]   (K padded 72->96, N padded 18->32)
    for (int e = t; e < NT2_ * KC2_ * 32 * 16; e += 256) {
        int f = e >> 9, rem = e & 511, lane = rem >> 4, i = rem & 15;
        int nt2 = f / KC2_, kc = f % KC2_;
        int nl = lane & 15, kh = lane >> 4;
        int k = kc * 32 + ((i < 8) ? (kh * 8 + i) : (16 + kh * 8 + (i - 8)));
        int n = nt2 * 16 + nl;
        float v = (k < HID_ && n < C_) ? w2[n * HID_ + k] : 0.0f;
        b2f[e] = (__bf16)v;
    }
    for (int e = t; e < B_ * C_; e += 256) {
        gsum[e] = 0.0f;
        gmax[e] = -__builtin_inff();
    }
}

// ---------------------------------------------------------------------------
// k1: dwconv + LN + MLP(GELU) + layer-scale, y -> ws, per-(b,c) sum/max.
// Block: 256 threads (8 waves), 256 consecutive L positions of one batch.
// ---------------------------------------------------------------------------
__global__ __launch_bounds__(256)
void main_kernel(const float* __restrict__ x,
                 const float* __restrict__ dw_w, const float* __restrict__ dw_b,
                 const float* __restrict__ ln_w, const float* __restrict__ ln_b,
                 const float* __restrict__ b1,   const float* __restrict__ b2,
                 const float* __restrict__ gamma,
                 const __bf16* __restrict__ b1f, const __bf16* __restrict__ b2f,
                 float* __restrict__ yws, float* __restrict__ gsum, float* __restrict__ gmax)
{
    __shared__ __align__(16) __bf16 hL[256][32];      // LN output, K padded to 32 (16 KB)
    __shared__ __align__(16) __bf16 gL[8][16][96];    // per-wave GELU staging, K padded to 96 (24 KB)
    __shared__ float sumL[C_];
    __shared__ int   maxL[C_];

    const int t    = threadIdx.x;
    const int b    = blockIdx.y;
    const int l0   = blockIdx.x * 256;
    const int lane = t & 31;
    const int wave = t >> 5;
    const int ln15 = lane & 15;        // A: row M, B/D: column N
    const int kh   = lane >> 4;        // lane half selects K/M sub-range

    if (t < C_) { sumL[t] = 0.0f; maxL[t] = (int)0xff800000; }
    // zero staging LDS once: columns 72..95 stay zero forever (K padding)
    for (int i = t; i < 8 * 16 * 96; i += 256) ((__bf16*)gL)[i] = (__bf16)0.0f;

    // ---- phase 1: conv + LN, one position per thread (interior fast path)
    if (l0 >= 3 && l0 + 256 + 3 <= L_)
        conv_ln_row<false>(x, dw_w, dw_b, ln_w, ln_b, b, l0 + t, &hL[t][0]);
    else
        conv_ln_row<true >(x, dw_w, dw_b, ln_w, ln_b, b, l0 + t, &hL[t][0]);
    __syncthreads();

    // ---- load packed weight fragments (uniform across waves, served by L2/L0)
    Frag32B w1f[NT1_];
    #pragma unroll
    for (int nt = 0; nt < NT1_; ++nt) {
        const uint4* p = (const uint4*)(b1f + (size_t)(nt * 32 + lane) * 16);
        w1f[nt].u[0] = p[0]; w1f[nt].u[1] = p[1];
    }
    Frag32B w2f[NT2_][KC2_];
    #pragma unroll
    for (int nt2 = 0; nt2 < NT2_; ++nt2)
        #pragma unroll
        for (int kc = 0; kc < KC2_; ++kc) {
            const uint4* p = (const uint4*)(b2f + (size_t)((nt2 * KC2_ + kc) * 32 + lane) * 16);
            w2f[nt2][kc].u[0] = p[0]; w2f[nt2][kc].u[1] = p[1];
        }

    // ---- phase 2: each wave runs two 16-position M-tiles through the MLP
    for (int mt = 0; mt < 2; ++mt) {
        const int pbase = wave * 32 + mt * 16;   // tile base within the 256 positions

        // A fragment (16x32 bf16): lane row M=ln15; elems 0..7 -> K=kh*8+i, 8..15 -> K=16+kh*8+i
        Frag32B a;
        a.u[0] = *(const uint4*)&hL[pbase + ln15][kh * 8];
        a.u[1] = *(const uint4*)&hL[pbase + ln15][16 + kh * 8];

        // GEMM1: (16x32) x (32x16) x 5 N-tiles, f32 accumulate
        v8f acc[NT1_];
        #pragma unroll
        for (int nt = 0; nt < NT1_; ++nt) {
            v8f z = {};
            acc[nt] = __builtin_amdgcn_wmma_f32_16x16x32_bf16(
                false, a.v, false, w1f[nt].v, (short)0, z, false, false);
        }

        __syncthreads();   // previous tile's staging reads are done everywhere

        // bias + exact GELU -> bf16 staging (D layout: row M = r + 8*kh, col N = ln15)
        #pragma unroll
        for (int nt = 0; nt < NT1_; ++nt) {
            int n = nt * 16 + ln15;
            if (n < HID_) {
                float bb = b1[n];
                #pragma unroll
                for (int r = 0; r < 8; ++r)
                    gL[wave][r + 8 * kh][n] = (__bf16)gelu_exact(acc[nt][r] + bb);
            }
        }
        __syncthreads();

        // GEMM2: (16x96) x (96x16) x 2 N-tiles over 3 K-chunks
        v8f acc2[NT2_];
        #pragma unroll
        for (int nt2 = 0; nt2 < NT2_; ++nt2) { v8f z = {}; acc2[nt2] = z; }
        #pragma unroll
        for (int kc = 0; kc < KC2_; ++kc) {
            Frag32B a2;
            a2.u[0] = *(const uint4*)&gL[wave][ln15][kc * 32 + kh * 8];
            a2.u[1] = *(const uint4*)&gL[wave][ln15][kc * 32 + 16 + kh * 8];
            #pragma unroll
            for (int nt2 = 0; nt2 < NT2_; ++nt2)
                acc2[nt2] = __builtin_amdgcn_wmma_f32_16x16x32_bf16(
                    false, a2.v, false, w2f[nt2][kc].v, (short)0, acc2[nt2], false, false);
        }

        // bias2 + gamma, store y (8 consecutive positions/lane -> 2x b128), reduce sum/max
        #pragma unroll
        for (int nt2 = 0; nt2 < NT2_; ++nt2) {
            int n = nt2 * 16 + ln15;
            if (n < C_) {
                float bb = b2[n], gm = gamma[n];
                float v0 = (acc2[nt2][0] + bb) * gm;
                float v1 = (acc2[nt2][1] + bb) * gm;
                float v2 = (acc2[nt2][2] + bb) * gm;
                float v3 = (acc2[nt2][3] + bb) * gm;
                float v4 = (acc2[nt2][4] + bb) * gm;
                float v5 = (acc2[nt2][5] + bb) * gm;
                float v6 = (acc2[nt2][6] + bb) * gm;
                float v7 = (acc2[nt2][7] + bb) * gm;
                size_t off = ((size_t)b * C_ + n) * L_ + (size_t)(l0 + pbase + 8 * kh);
                float4* p = (float4*)(yws + off);
                p[0] = make_float4(v0, v1, v2, v3);
                p[1] = make_float4(v4, v5, v6, v7);
                float ls = ((v0 + v1) + (v2 + v3)) + ((v4 + v5) + (v6 + v7));
                float lm = fmaxf(fmaxf(fmaxf(v0, v1), fmaxf(v2, v3)),
                                 fmaxf(fmaxf(v4, v5), fmaxf(v6, v7)));
                atomicAdd(&sumL[n], ls);
                atomicMaxFloatShared(&maxL[n], lm);
            }
        }
        __syncthreads();
    }

    if (t < C_) {
        atomicAdd(&gsum[b * C_ + t], sumL[t]);
        atomicMaxFloatGlobal(&gmax[b * C_ + t], __int_as_float(maxL[t]));
    }
}

// ---------------------------------------------------------------------------
// k2: CBAM channel attention (bottleneck Cb = 1): att = sigmoid(fc(avg)+fc(max))
// ---------------------------------------------------------------------------
__global__ void attention_kernel(const float* __restrict__ gsum, const float* __restrict__ gmax,
                                 const float* __restrict__ caw1, const float* __restrict__ caw2,
                                 float* __restrict__ att)
{
    const int b = threadIdx.x;
    if (b < B_) {
        const float invL = 1.0f / (float)L_;
        float sa = 0.0f, sm = 0.0f;
        #pragma unroll
        for (int c = 0; c < C_; ++c) {
            sa = fmaf(gsum[b * C_ + c] * invL, caw1[c], sa);
            sm = fmaf(gmax[b * C_ + c],        caw1[c], sm);
        }
        float r = fmaxf(sa, 0.0f) + fmaxf(sm, 0.0f);   // relu(hidden) shared-MLP, summed
        #pragma unroll
        for (int c = 0; c < C_; ++c) {
            float z = r * caw2[c];
            att[b * C_ + c] = __builtin_amdgcn_rcpf(1.0f + __expf(-z));
        }
    }
}

// ---------------------------------------------------------------------------
// k3: out = gelu(y * att + x), float4-vectorized stream (y served from L2)
// ---------------------------------------------------------------------------
__global__ __launch_bounds__(256)
void epilogue_kernel(const float4* __restrict__ x4, const float4* __restrict__ y4,
                     const float* __restrict__ att, float4* __restrict__ out4)
{
    const int l4 = L_ / 4;
    const int n4 = B_ * C_ * l4;
    for (int i = blockIdx.x * blockDim.x + threadIdx.x; i < n4;
         i += gridDim.x * blockDim.x) {
        int bc = i / l4;
        float a  = att[bc];
        float4 xv = x4[i];
        float4 yv = y4[i];
        float4 o;
        o.x = gelu_exact(fmaf(yv.x, a, xv.x));
        o.y = gelu_exact(fmaf(yv.y, a, xv.y));
        o.z = gelu_exact(fmaf(yv.z, a, xv.z));
        o.w = gelu_exact(fmaf(yv.w, a, xv.w));
        out4[i] = o;
    }
}

// ---------------------------------------------------------------------------
extern "C" void kernel_launch(void* const* d_in, const int* in_sizes, int n_in,
                              void* d_out, int out_size, void* d_ws, size_t ws_size,
                              hipStream_t stream)
{
    (void)in_sizes; (void)n_in; (void)out_size; (void)ws_size;

    const float* x     = (const float*)d_in[0];
    const float* dw_w  = (const float*)d_in[1];
    const float* dw_b  = (const float*)d_in[2];
    const float* ln_w  = (const float*)d_in[3];
    const float* ln_b  = (const float*)d_in[4];
    const float* w1    = (const float*)d_in[5];
    const float* b1    = (const float*)d_in[6];
    const float* w2    = (const float*)d_in[7];
    const float* b2    = (const float*)d_in[8];
    const float* gamma = (const float*)d_in[9];
    const float* caw1  = (const float*)d_in[10];
    const float* caw2  = (const float*)d_in[11];
    float* out = (float*)d_out;

    char* ws = (char*)d_ws;
    size_t o = 0;
    float*  yws  = (float*)(ws + o);  o += (size_t)B_ * C_ * L_ * sizeof(float);
    float*  gsum = (float*)(ws + o);  o += 1536;            // 288 floats, padded
    float*  gmax = (float*)(ws + o);  o += 1536;
    float*  att  = (float*)(ws + o);  o += 1536;
    __bf16* b1f  = (__bf16*)(ws + o); o += (size_t)NT1_ * 32 * 16 * 2 + 256;
    __bf16* b2f  = (__bf16*)(ws + o); o += (size_t)NT2_ * KC2_ * 32 * 16 * 2 + 256;

    pack_init_kernel<<<1, 256, 0, stream>>>(w1, w2, b1f, b2f, gsum, gmax);

    dim3 grid1(L_ / 256, B_);
    main_kernel<<<grid1, 256, 0, stream>>>(x, dw_w, dw_b, ln_w, ln_b, b1, b2, gamma,
                                           b1f, b2f, yws, gsum, gmax);

    attention_kernel<<<1, 32, 0, stream>>>(gsum, gmax, caw1, caw2, att);

    epilogue_kernel<<<4096, 256, 0, stream>>>((const float4*)x, (const float4*)yws,
                                              att, (float4*)out);
}
